// Transformer_34308198760755
// MI455X (gfx1250) — matrix operand order repaired and checked by reading, not compiled, not run on hardware
//
#include <hip/hip_runtime.h>
#include <math.h>

// ---------------------------------------------------------------------------
// Deformable-DETR decoder layer for MI455X (gfx1250, wave32, WMMA).
// All GEMMs run on v_wmma_f32_16x16x32_bf16 (bf16 in / f32 accumulate).
// Invariants for every launch: K % 32 == 0, aRow % 4 == 0, A/B bases 16-B
// aligned (score buffers are K-padded; softmax zero-fills the pad).
// ---------------------------------------------------------------------------

typedef __attribute__((ext_vector_type(16))) __bf16 v16bf;
typedef __attribute__((ext_vector_type(8)))  __bf16 v8bf;
typedef __attribute__((ext_vector_type(4)))  __bf16 v4bf;
typedef __attribute__((ext_vector_type(8)))  float  v8f;

#define BM 128
#define BN 64
#define BK 32
#define AST 40   // A tile LDS row stride (bf16): 80 B rows -> 16-B aligned chunks
#define BKP 40   // B tile (transposed) LDS row stride (bf16): 80 B rows

// Generic batched GEMM:  C[z] = act(scale * A[z] @ B[z] + bias)
//   z = zBase + blockIdx.z ; zb = z/nh ; zh = z%nh
//   A element (m,k): A + zb*aSB + zh*aSH + m*aRow + k
//   B element (k,n): B + zb*bSB + zh*bSH + k*bK  + n*bN   (bN=ld => transposed W)
//   C element (m,n): C + zb*cSB + zh*cSH + m*cRow + n
__global__ __launch_bounds__(256) void gemm_bf16_wmma(
    const float* __restrict__ A, long long aSB, long long aSH, long long aRow,
    const float* __restrict__ Bm, long long bSB, long long bSH, long long bK, long long bN,
    const float* __restrict__ bias,
    float* __restrict__ C, long long cSB, long long cSH, long long cRow,
    int M, int N, int K, int nh, int zBase, float scale, int act)
{
  __shared__ __bf16 As[BM * AST];    // [row][k]  (k-contiguous)
  __shared__ __bf16 BsT[BN * BKP];   // [col][k]  (k-contiguous, transposed)

  const int z  = zBase + (int)blockIdx.z;
  const int zb = z / nh, zh = z % nh;
  const float* Ab = A  + (long long)zb * aSB + (long long)zh * aSH;
  const float* Bb = Bm + (long long)zb * bSB + (long long)zh * bSH;
  float*       Cb = C  + (long long)zb * cSB + (long long)zh * cSH;

  const int m0  = (int)blockIdx.x * BM;
  const int n0  = (int)blockIdx.y * BN;
  const int tid = (int)threadIdx.x;
  const int wave = tid >> 5, lane = tid & 31;
  const int wm = (wave >> 1) * 32;   // 4 waves along M
  const int wn = (wave & 1) * 32;    // 2 waves along N

  // A loader: 1 row per thread (128 rows x 2 threads/row covering 16 k each)
  const int ar_row = tid >> 1;
  const int ar_q   = (tid & 1) * 16;
  // B loader: 64 cols x 4 k-groups of 8 contiguous k
  const int bn_n = tid & 63;
  const int bk_g = (tid >> 6) * 8;

  // clamped base addresses (duplicate rows/cols are never stored by epilogue)
  const int gmc = (m0 + ar_row) < M ? (m0 + ar_row) : (M - 1);
  const int gnc = (n0 + bn_n)  < N ? (n0 + bn_n)  : (N - 1);
  const float* aBase = Ab + (long long)gmc * aRow + ar_q;
  const float* bBase = Bb + (long long)gnc * bN;

  v8f acc[2][2] = {};
  float4 ar[4];
  float  br[8];

  // stage tile k0=0 into registers
  {
    const float4* ap = reinterpret_cast<const float4*>(aBase);
    #pragma unroll
    for (int i = 0; i < 4; ++i) ar[i] = ap[i];
    #pragma unroll
    for (int i = 0; i < 8; ++i) br[i] = bBase[(long long)(bk_g + i) * bK];
  }

  for (int k0 = 0; k0 < K; k0 += BK) {
    // ---- commit staged registers to LDS (convert f32 -> bf16) ----
    {
      __bf16* adst = &As[ar_row * AST + ar_q];
      #pragma unroll
      for (int i = 0; i < 4; ++i) {
        v4bf h;
        h[0] = (__bf16)ar[i].x; h[1] = (__bf16)ar[i].y;
        h[2] = (__bf16)ar[i].z; h[3] = (__bf16)ar[i].w;
        *reinterpret_cast<v4bf*>(adst + i * 4) = h;
      }
      __bf16* bdst = &BsT[bn_n * BKP + bk_g];
      #pragma unroll
      for (int half = 0; half < 2; ++half) {
        v4bf h;
        #pragma unroll
        for (int i = 0; i < 4; ++i) h[i] = (__bf16)br[half * 4 + i];
        *reinterpret_cast<v4bf*>(bdst + half * 4) = h;
      }
    }
    __syncthreads();

    // ---- issue next tile's global loads (overlap with WMMA below) ----
    if (k0 + BK < K) {
      const float4* ap = reinterpret_cast<const float4*>(aBase + k0 + BK);
      #pragma unroll
      for (int i = 0; i < 4; ++i) ar[i] = ap[i];
      const float* bp = bBase + (long long)(k0 + BK) * bK;
      #pragma unroll
      for (int i = 0; i < 8; ++i) br[i] = bp[(long long)(bk_g + i) * bK];
      if (k0 + 2 * BK < K)  // speculative prefetch two tiles ahead (global_prefetch_b8)
        __builtin_prefetch(aBase + k0 + 2 * BK, 0, 1);
    }

    // ---- fragments: aligned 16-B LDS chunks, concat via shufflevector ----
    const int arow = lane & 15;
    const int akb  = (lane >> 4) << 3;   // A: lanes 0-15 K 0..7/16..23 ; 16-31 K 8..15/24..31
    const int bcol = lane & 15;
    const int bkb  = (lane >> 4) << 3;   // B: lanes 0-15 K 0..7/16..23 ; 16-31 K 8..15/24..31
    v16bf af[2], bfr[2];
    #pragma unroll
    for (int i = 0; i < 2; ++i) {
      const __bf16* p = &As[(wm + i * 16 + arow) * AST];
      v8bf lo = *reinterpret_cast<const v8bf*>(p + akb);
      v8bf hi = *reinterpret_cast<const v8bf*>(p + akb + 16);
      af[i] = __builtin_shufflevector(lo, hi, 0,1,2,3,4,5,6,7,8,9,10,11,12,13,14,15);
    }
    #pragma unroll
    for (int i = 0; i < 2; ++i) {
      const __bf16* p = &BsT[(wn + i * 16 + bcol) * BKP + bkb];
      v8bf lo = *reinterpret_cast<const v8bf*>(p);
      v8bf hi = *reinterpret_cast<const v8bf*>(p + 16);
      bfr[i] = __builtin_shufflevector(lo, hi, 0,1,2,3,4,5,6,7,8,9,10,11,12,13,14,15);
    }
    #pragma unroll
    for (int i = 0; i < 2; ++i)
      #pragma unroll
      for (int j = 0; j < 2; ++j)
        acc[i][j] = __builtin_amdgcn_wmma_f32_16x16x32_bf16(
            false, af[i], false, bfr[j], (short)0, acc[i][j], false, false);
    __syncthreads();
  }

  // ---- epilogue: C layout VGPR r -> (M = r or 8+r), N = lane&15 ----
  const int cn = lane & 15;
  const int cm = (lane >> 4) << 3;
  #pragma unroll
  for (int i = 0; i < 2; ++i)
    #pragma unroll
    for (int j = 0; j < 2; ++j) {
      int nIdx = n0 + wn + j * 16 + cn;
      if (nIdx >= N) continue;
      float bb = bias ? bias[nIdx] : 0.0f;
      #pragma unroll
      for (int r = 0; r < 8; ++r) {
        int mIdx = m0 + wm + i * 16 + cm + r;
        if (mIdx < M) {
          float v = acc[i][j][r] * scale + bb;
          if (act) v = fmaxf(v, 0.0f);
          Cb[(long long)mIdx * cRow + nIdx] = v;
        }
      }
    }
}

// ---------------------------------------------------------------------------
__global__ __launch_bounds__(256) void add_kernel(const float* __restrict__ a,
                                                  const float* __restrict__ b,
                                                  float* __restrict__ o, long long n)
{
  long long i = (long long)blockIdx.x * 256 + threadIdx.x;
  if (i < n) o[i] = a[i] + b[i];
}

// softmax over rows (one wave32 per row); zero-fills pad columns [N, zeroTo)
__global__ __launch_bounds__(256) void softmax_rows(float* __restrict__ p,
                                                    int nRows, int N, int ld, int zeroTo)
{
  int row  = (int)blockIdx.x * 8 + ((int)threadIdx.x >> 5);
  int lane = (int)threadIdx.x & 31;
  if (row >= nRows) return;
  float* r = p + (long long)row * ld;
  float mx = -3.0e38f;
  for (int j = lane; j < N; j += 32) mx = fmaxf(mx, r[j]);
  for (int o = 16; o; o >>= 1) mx = fmaxf(mx, __shfl_xor(mx, o, 32));
  float sum = 0.0f;
  for (int j = lane; j < N; j += 32) { float e = __expf(r[j] - mx); r[j] = e; sum += e; }
  for (int o = 16; o; o >>= 1) sum += __shfl_xor(sum, o, 32);
  float inv = 1.0f / sum;
  for (int j = lane; j < N; j += 32) r[j] *= inv;
  for (int j = N + lane; j < zeroTo; j += 32) r[j] = 0.0f;
}

// out = LN(x + y) * g + b   (D=256, one wave32 per row, 8 elems/lane)
__global__ __launch_bounds__(256) void ln_res(const float* __restrict__ x,
                                              const float* __restrict__ y,
                                              const float* __restrict__ g,
                                              const float* __restrict__ b,
                                              float* __restrict__ out, int rows)
{
  int row  = (int)blockIdx.x * 8 + ((int)threadIdx.x >> 5);
  int lane = (int)threadIdx.x & 31;
  if (row >= rows) return;
  const float* xp = x + (long long)row * 256;
  const float* yp = y + (long long)row * 256;
  float v[8], s = 0.0f;
  #pragma unroll
  for (int i = 0; i < 8; ++i) { int c = i * 32 + lane; float t = xp[c] + yp[c]; v[i] = t; s += t; }
  for (int o = 16; o; o >>= 1) s += __shfl_xor(s, o, 32);
  float mean = s * (1.0f / 256.0f);
  float var = 0.0f;
  #pragma unroll
  for (int i = 0; i < 8; ++i) { float d = v[i] - mean; var += d * d; }
  for (int o = 16; o; o >>= 1) var += __shfl_xor(var, o, 32);
  float inv = rsqrtf(var * (1.0f / 256.0f) + 1e-5f);
  float* op = out + (long long)row * 256;
  #pragma unroll
  for (int i = 0; i < 8; ++i) { int c = i * 32 + lane; op[c] = (v[i] - mean) * inv * g[c] + b[c]; }
}

// Multi-scale deformable bilinear sampling.
// One wave32 per (row=q*8+b, head): lane = head-dim channel.
__global__ __launch_bounds__(256) void deform_sample(
    const float* __restrict__ ref,   // (NQ,BS,NL,4)
    const float* __restrict__ off,   // rows x 256 : (h,l,p,2)
    const float* __restrict__ aw,    // rows x 128 : (h,l,p) softmaxed over 16
    const float* __restrict__ v,     // (MEM*BS, 256)
    float* __restrict__ cbuf)        // rows x 256
{
  const int Hs[4] = {92, 46, 23, 12};
  const int Ws[4] = {160, 80, 40, 20};
  const int Ss[4] = {0, 14720, 18400, 19320};

  int gw   = (int)blockIdx.x * 8 + ((int)threadIdx.x >> 5);
  int lane = (int)threadIdx.x & 31;
  int h    = gw & 7;
  int row  = gw >> 3;
  if (row >= 7200) return;
  int b = row & 7;

  const float* refp = ref + (long long)row * 16;
  const float* offp = off + (long long)row * 256 + h * 32;
  const float* awp  = aw  + (long long)row * 128 + h * 16;

  float acc = 0.0f;
  #pragma unroll
  for (int l = 0; l < 4; ++l) {
    const int Hl = Hs[l], Wl = Ws[l], s = Ss[l];
    float cx = refp[l * 4 + 0], cy = refp[l * 4 + 1];
    float pw = refp[l * 4 + 2], ph = refp[l * 4 + 3];
    #pragma unroll
    for (int p = 0; p < 4; ++p) {
      float ox = offp[(l * 4 + p) * 2 + 0];
      float oy = offp[(l * 4 + p) * 2 + 1];
      float x = (cx + ox * 0.25f * pw * 0.5f) * (float)Wl - 0.5f;
      float y = (cy + oy * 0.25f * ph * 0.5f) * (float)Hl - 0.5f;
      float x0f = floorf(x), y0f = floorf(y);
      float fx = x - x0f, fy = y - y0f;
      int x0 = (int)x0f, y0 = (int)y0f;
      float w = awp[l * 4 + p];
      float sample = 0.0f;
      #pragma unroll
      for (int dy = 0; dy < 2; ++dy)
        #pragma unroll
        for (int dx = 0; dx < 2; ++dx) {
          int xi = x0 + dx, yi = y0 + dy;
          float ww = (dx ? fx : 1.0f - fx) * (dy ? fy : 1.0f - fy);
          bool valid = (xi >= 0) & (xi < Wl) & (yi >= 0) & (yi < Hl);
          int xc = min(max(xi, 0), Wl - 1);
          int yc = min(max(yi, 0), Hl - 1);
          long long vrow = (long long)(s + yc * Wl + xc) * 8 + b;
          float samp = v[vrow * 256 + h * 32 + lane];
          sample += valid ? ww * samp : 0.0f;
        }
      acc += sample * w;
    }
  }
  cbuf[(long long)row * 256 + h * 32 + lane] = acc;
}

// ---------------------------------------------------------------------------
extern "C" void kernel_launch(void* const* d_in, const int* in_sizes, int n_in,
                              void* d_out, int out_size, void* d_ws, size_t ws_size,
                              hipStream_t stream)
{
  constexpr int Dm = 256, FF = 1024, R = 7200;         // R = NQ*BS
  constexpr int MR = 19560 * 8;                        // memory rows
  constexpr long long RD = (long long)R * Dm;

  const float* tgt_in    = (const float*)d_in[0];
  const float* pos       = (const float*)d_in[1];
  const float* ref       = (const float*)d_in[2];
  const float* memory    = (const float*)d_in[3];
  // d_in[4] spatial_shapes, d_in[5] level_start_index, d_in[6] num: module constants, hardcoded.
  const float* Wo = (const float*)d_in[7];  const float* bo = (const float*)d_in[8];
  const float* Wa = (const float*)d_in[9];  const float* ba = (const float*)d_in[10];
  const float* Wv = (const float*)d_in[11]; const float* bv = (const float*)d_in[12];
  const float* Wm = (const float*)d_in[13]; const float* bm = (const float*)d_in[14];
  const float* upd_in_w  = (const float*)d_in[15]; const float* upd_in_b  = (const float*)d_in[16];
  const float* upd_out_w = (const float*)d_in[17]; const float* upd_out_b = (const float*)d_in[18];
  const float* n4_g = (const float*)d_in[19]; const float* n4_b = (const float*)d_in[20];
  const float* sa_in_w  = (const float*)d_in[21]; const float* sa_in_b  = (const float*)d_in[22];
  const float* sa_out_w = (const float*)d_in[23]; const float* sa_out_b = (const float*)d_in[24];
  const float* n2_g = (const float*)d_in[25]; const float* n2_b = (const float*)d_in[26];
  const float* n1_g = (const float*)d_in[27]; const float* n1_b = (const float*)d_in[28];
  const float* l1_w = (const float*)d_in[29]; const float* l1_b = (const float*)d_in[30];
  const float* l2_w = (const float*)d_in[31]; const float* l2_b = (const float*)d_in[32];
  const float* n3_g = (const float*)d_in[33]; const float* n3_b = (const float*)d_in[34];
  const float* l3_w = (const float*)d_in[35]; const float* l3_b = (const float*)d_in[36];
  const float* l4_w = (const float*)d_in[37]; const float* l4_b = (const float*)d_in[38];
  const float* n5_g = (const float*)d_in[39]; const float* n5_b = (const float*)d_in[40];

  // ---- workspace layout (floats) ----
  constexpr long long VPPAD = 57600;                   // vp tail pad for padded-K PV reads
  constexpr long long AWSZ  = (long long)R * 128;
  constexpr long long FFSZ  = (long long)R * FF;
  constexpr long long S1SZ  = 64LL * 100 * 128;        // scores1, K-padded to 128
  constexpr long long S2SZ  = 8LL * 900 * 928;         // scores2 (one batch), K-padded to 928
  constexpr long long VSZ   = (long long)MR * Dm;
  constexpr long long TOTAL = 9 * RD + VPPAD + AWSZ + FFSZ + S1SZ + S2SZ + VSZ;
  if (ws_size < (size_t)(TOTAL * sizeof(float))) return;

  float* ws = (float*)d_ws;
  float* tgt_cur = ws;                 // R x 256 (mutable tgt)
  float* qk   = tgt_cur + RD;
  float* qp   = qk   + RD;
  float* kp   = qp   + RD;
  float* vp   = kp   + RD;             // R x 256 (+ VPPAD tail)
  float* obuf = vp   + RD + VPPAD;
  float* ptmp = obuf + RD;
  float* offb = ptmp + RD;
  float* cbuf = offb + RD;
  float* awb  = cbuf + RD;             // R x 128
  float* ffn  = awb  + AWSZ;           // R x 1024
  float* s1   = ffn  + FFSZ;           // 64 x 100 x 128
  float* s2   = s1   + S1SZ;           // 8 x 900 x 928
  float* vbig = s2   + S2SZ;           // MR x 256

  auto gemm = [&](const float* A, long long aSB, long long aSH, long long aRow,
                  const float* B, long long bSB, long long bSH, long long bK, long long bN,
                  const float* bias, float* C, long long cSB, long long cSH, long long cRow,
                  int M, int N, int K, int nh, int zBase, int nz, float scale, int act) {
    dim3 g((M + BM - 1) / BM, (N + BN - 1) / BN, nz);
    gemm_bf16_wmma<<<g, 256, 0, stream>>>(A, aSB, aSH, aRow, B, bSB, bSH, bK, bN,
                                          bias, C, cSB, cSH, cRow, M, N, K, nh, zBase, scale, act);
  };
  const float SC = 0.17677669529f;     // 1/sqrt(32)
  const long long TAIL = 6400;         // num*BS

  // tgt_cur = tgt ; qk = tgt + pos
  hipMemcpyAsync(tgt_cur, tgt_in, RD * sizeof(float), hipMemcpyDeviceToDevice, stream);
  add_kernel<<<(int)((RD + 255) / 256), 256, 0, stream>>>(tgt_cur, pos, qk, RD);

  // ---- MHA #1 over tail 100 queries ----
  gemm(qk + TAIL * Dm, 0, 0, Dm, upd_in_w, 0, 0, 1, Dm, upd_in_b,
       qp + TAIL * Dm, 0, 0, Dm, 800, Dm, Dm, 1, 0, 1, 1.0f, 0);
  gemm(qk + TAIL * Dm, 0, 0, Dm, upd_in_w + Dm * Dm, 0, 0, 1, Dm, upd_in_b + Dm,
       kp + TAIL * Dm, 0, 0, Dm, 800, Dm, Dm, 1, 0, 1, 1.0f, 0);
  gemm(tgt_cur + TAIL * Dm, 0, 0, Dm, upd_in_w + 2 * Dm * Dm, 0, 0, 1, Dm, upd_in_b + 2 * Dm,
       vp + TAIL * Dm, 0, 0, Dm, 800, Dm, Dm, 1, 0, 1, 1.0f, 0);
  gemm(qp + TAIL * Dm, 256, 32, 2048, kp + TAIL * Dm, 256, 32, 1, 2048, nullptr,
       s1, 102400, 12800, 128, 100, 100, 32, 8, 0, 64, SC, 0);
  softmax_rows<<<(6400 + 7) / 8, 256, 0, stream>>>(s1, 6400, 100, 128, 128);
  gemm(s1, 102400, 12800, 128, vp + TAIL * Dm, 256, 32, 2048, 1, nullptr,
       obuf + TAIL * Dm, 256, 32, 2048, 100, 32, 128, 8, 0, 64, 1.0f, 0);
  gemm(obuf + TAIL * Dm, 0, 0, Dm, upd_out_w, 0, 0, 1, Dm, upd_out_b,
       ptmp + TAIL * Dm, 0, 0, Dm, 800, Dm, Dm, 1, 0, 1, 1.0f, 0);
  ln_res<<<(800 + 7) / 8, 256, 0, stream>>>(tgt_cur + TAIL * Dm, ptmp + TAIL * Dm,
                                            n4_g, n4_b, tgt_cur + TAIL * Dm, 800);

  // ---- MHA #2 over all 900 queries ----
  add_kernel<<<(int)((RD + 255) / 256), 256, 0, stream>>>(tgt_cur, pos, qk, RD);
  gemm(qk, 0, 0, Dm, sa_in_w, 0, 0, 1, Dm, sa_in_b, qp, 0, 0, Dm, R, Dm, Dm, 1, 0, 1, 1.0f, 0);
  gemm(qk, 0, 0, Dm, sa_in_w + Dm * Dm, 0, 0, 1, Dm, sa_in_b + Dm, kp, 0, 0, Dm, R, Dm, Dm, 1, 0, 1, 1.0f, 0);
  gemm(tgt_cur, 0, 0, Dm, sa_in_w + 2 * Dm * Dm, 0, 0, 1, Dm, sa_in_b + 2 * Dm, vp, 0, 0, Dm, R, Dm, Dm, 1, 0, 1, 1.0f, 0);
  for (int bc = 0; bc < 8; ++bc) {      // chunk scores per batch to bound workspace
    gemm(qp, 256, 32, 2048, kp, 256, 32, 1, 2048, nullptr,
         s2, 0, 835200, 928, 900, 900, 32, 8, bc * 8, 8, SC, 0);
    softmax_rows<<<(7200 + 7) / 8, 256, 0, stream>>>(s2, 7200, 900, 928, 928);
    gemm(s2, 0, 835200, 928, vp, 256, 32, 2048, 1, nullptr,
         obuf, 256, 32, 2048, 900, 32, 928, 8, bc * 8, 8, 1.0f, 0);
  }
  gemm(obuf, 0, 0, Dm, sa_out_w, 0, 0, 1, Dm, sa_out_b, ptmp, 0, 0, Dm, R, Dm, Dm, 1, 0, 1, 1.0f, 0);
  ln_res<<<(R + 7) / 8, 256, 0, stream>>>(tgt_cur, ptmp, n2_g, n2_b, tgt_cur, R);

  // ---- deformable attention ----
  add_kernel<<<(int)((RD + 255) / 256), 256, 0, stream>>>(tgt_cur, pos, qk, RD);
  gemm(memory, 0, 0, Dm, Wv, 0, 0, Dm, 1, bv, vbig, 0, 0, Dm, MR, Dm, Dm, 1, 0, 1, 1.0f, 0);
  gemm(qk, 0, 0, Dm, Wo, 0, 0, Dm, 1, bo, offb, 0, 0, Dm, R, Dm, Dm, 1, 0, 1, 1.0f, 0);
  gemm(qk, 0, 0, Dm, Wa, 0, 0, 128, 1, ba, awb, 0, 0, 128, R, 128, Dm, 1, 0, 1, 1.0f, 0);
  softmax_rows<<<(57600 + 7) / 8, 256, 0, stream>>>(awb, 57600, 16, 16, 16);
  deform_sample<<<R, 256, 0, stream>>>(ref, offb, awb, vbig, cbuf);
  gemm(cbuf, 0, 0, Dm, Wm, 0, 0, Dm, 1, bm, ptmp, 0, 0, Dm, R, Dm, Dm, 1, 0, 1, 1.0f, 0);
  ln_res<<<(R + 7) / 8, 256, 0, stream>>>(tgt_cur, ptmp, n1_g, n1_b, tgt_cur, R);

  // ---- FFN branch 1 -> tgt_cti (d_out[0:RD]) ----
  float* out0 = (float*)d_out;
  float* out1 = out0 + RD;
  gemm(tgt_cur, 0, 0, Dm, l1_w, 0, 0, FF, 1, l1_b, ffn, 0, 0, FF, R, FF, Dm, 1, 0, 1, 1.0f, 1);
  gemm(ffn, 0, 0, FF, l2_w, 0, 0, Dm, 1, l2_b, ptmp, 0, 0, Dm, R, Dm, FF, 1, 0, 1, 1.0f, 0);
  ln_res<<<(R + 7) / 8, 256, 0, stream>>>(tgt_cur, ptmp, n3_g, n3_b, out0, R);

  // ---- FFN branch 2 -> tgt_aligned (d_out[RD:2RD]) ----
  gemm(tgt_cur, 0, 0, Dm, l3_w, 0, 0, FF, 1, l3_b, ffn, 0, 0, FF, R, FF, Dm, 1, 0, 1, 1.0f, 1);
  gemm(ffn, 0, 0, FF, l4_w, 0, 0, Dm, 1, l4_b, ptmp, 0, 0, Dm, R, Dm, FF, 1, 0, 1, 1.0f, 0);
  ln_res<<<(R + 7) / 8, 256, 0, stream>>>(tgt_cur, ptmp, n5_g, n5_b, out1, R);
}
